// ValueMulti_66005057405375
// MI455X (gfx1250) — matrix operand order repaired and checked by reading, not compiled, not run on hardware
//
#include <hip/hip_runtime.h>
#include <hip/hip_bf16.h>

#define B_ 4
#define N_LIG 150
#define N_PROT 550
#define N_ 700
#define H_ 128
#define HEADS 4
#define KPH 32
#define HPH 32
#define L_ 6
#define K_ 48
#define EDGE_CH 128
#define PROT_F 27
#define NUM_CLASSES 13
#define FLAT (N_ * H_)
#define LOG2F_ 0.6931471805599453f

typedef __bf16 bf16;
typedef __attribute__((ext_vector_type(16))) bf16 v16bf;
typedef __attribute__((ext_vector_type(8))) float v8f;

// ---------------- helpers ----------------

__device__ __forceinline__ float sspf(float x) {
  // softplus(x) - log(2), numerically stable
  float ax = fabsf(x);
  return fmaxf(x, 0.f) + __logf(1.f + __expf(-ax)) - LOG2F_;
}

// A tile 16x32 bf16 from LDS (row-major, leading dim ld).
// ISA layout: lanes 0-15 -> M=lane, K in {0..7,16..23}; lanes 16-31 -> M=lane-16, K in {8..15,24..31}
__device__ __forceinline__ v16bf load_A_bf(const bf16* base, int ld, int lane) {
  int m = lane & 15;
  int kb = (lane & 16) ? 8 : 0;
  const bf16* row = base + m * ld + kb;
  v16bf a;
#pragma unroll
  for (int j = 0; j < 8; ++j) a[j] = row[j];
#pragma unroll
  for (int j = 8; j < 16; ++j) a[j] = row[j + 8];
  return a;
}

// B tile (32x16, K x N) from an LDS image already stored TRANSPOSED: wt[in][out], stride ld.
// ISA layout: lanes 0-15 -> N=lane, K in {0..7,16..23}; lanes 16-31 -> N=lane-16, K in {8..15,24..31}
__device__ __forceinline__ v16bf load_B_bf(const bf16* wt, int ld, int k0, int n0, int lane) {
  const bf16* col = wt + (k0 + ((lane & 16) ? 8 : 0)) * ld + n0 + (lane & 15);
  v16bf b;
#pragma unroll
  for (int j = 0; j < 8; ++j) b[j] = col[j * ld];
#pragma unroll
  for (int j = 8; j < 16; ++j) b[j] = col[(j + 8) * ld];
  return b;
}

#define WMMA_BF16(A, Bm, C) \
  __builtin_amdgcn_wmma_f32_16x16x32_bf16(false, (A), false, (Bm), (short)0, (C), false, false)

// ---------------- kernels ----------------

__global__ void embed_kernel(const float* __restrict__ state, const float* __restrict__ ppv,
                             const float* __restrict__ w_prot, const float* __restrict__ b_prot,
                             const float* __restrict__ w_lig, const float* __restrict__ b_lig,
                             float* __restrict__ h, float* __restrict__ pos) {
  int bn = blockIdx.x;
  int b = bn / N_, n = bn % N_;
  int o = threadIdx.x;
  float acc;
  if (n < N_PROT) {
    const float* base = ppv + (size_t)(b * N_PROT + n) * (3 + PROT_F);
    acc = b_prot[o];
#pragma unroll
    for (int i = 0; i < PROT_F; ++i) acc += base[3 + i] * w_prot[o * PROT_F + i];
    if (o < 3) pos[bn * 3 + o] = base[o];
  } else {
    const float* st = state + (size_t)(b * N_LIG + (n - N_PROT)) * 4;
    int tcls = (int)st[3];
    acc = b_lig[o] + w_lig[o * NUM_CLASSES + tcls];
    if (o < 3) pos[bn * 3 + o] = st[o];
  }
  h[(size_t)bn * H_ + o] = acc;
}

__global__ void knn_kernel(const float* __restrict__ pos, int* __restrict__ nbr,
                           float* __restrict__ dist) {
  __shared__ float d2[N_];
  __shared__ float rmin[128];
  __shared__ int ridx[128];
  int bn = blockIdx.x;
  int b = bn / N_, n = bn % N_;
  int t = threadIdx.x;
  float px = pos[bn * 3], py = pos[bn * 3 + 1], pz = pos[bn * 3 + 2];
  for (int j = t; j < N_; j += 128) {
    const float* q = pos + (size_t)(b * N_ + j) * 3;
    float dx = q[0] - px, dy = q[1] - py, dz = q[2] - pz;
    float d = dx * dx + dy * dy + dz * dz;
    if (j == n) d = 1.0e9f;
    d2[j] = d;
  }
  __syncthreads();
  for (int s = 0; s < K_; ++s) {
    float mv = 3.0e38f;
    int mi = 0;
    for (int j = t; j < N_; j += 128) {
      float v = d2[j];
      if (v < mv) { mv = v; mi = j; }
    }
    rmin[t] = mv; ridx[t] = mi;
    __syncthreads();
    for (int w = 64; w > 0; w >>= 1) {
      if (t < w && rmin[t + w] < rmin[t]) { rmin[t] = rmin[t + w]; ridx[t] = ridx[t + w]; }
      __syncthreads();
    }
    if (t == 0) {
      int j = ridx[0];
      nbr[bn * K_ + s] = j;
      dist[bn * K_ + s] = sqrtf(rmin[0]);
      d2[j] = 3.0e38f;
    }
    __syncthreads();
  }
}

__global__ void edge_kernel(const float* __restrict__ dist, unsigned short* __restrict__ edge_u16) {
  bf16* edge = reinterpret_cast<bf16*>(edge_u16);
  int bn = blockIdx.x;
  int c = threadIdx.x;
  const float step = 10.0f / (EDGE_CH - 1);
  const float coeff = -0.5f / (step * step);
  float off = c * step;
  for (int k = 0; k < K_; ++k) {
    float d = dist[bn * K_ + k] - off;
    float e = __expf(coeff * d * d);
    edge[(size_t)(bn * K_ + k) * EDGE_CH + c] = (bf16)e;
  }
}

// q/k/v projections for 16 nodes per block, all through WMMA.
// Three separate branchless tile loops so store targets are compile-time fixed.
__global__ __launch_bounds__(128) void qkv16_kernel(
    const float* __restrict__ h, const float* __restrict__ wq_l,
    const float* __restrict__ wk_l, const float* __restrict__ wv_l, float* __restrict__ qg,
    unsigned short* __restrict__ kx_u16, unsigned short* __restrict__ vx_u16) {
  __shared__ bf16 s_hb[16 * H_];
  __shared__ bf16 s_wqt[HEADS * KPH * HPH];  // transposed per head: [in][out]
  __shared__ bf16 s_wkt[HEADS * KPH * HPH];
  __shared__ bf16 s_wvt[HEADS * HPH * HPH];
  bf16* kx = reinterpret_cast<bf16*>(kx_u16);
  bf16* vx = reinterpret_cast<bf16*>(vx_u16);
  int t = threadIdx.x, lane = t & 31, wave = t >> 5;
  size_t base = (size_t)blockIdx.x * 16 * H_;
  for (int i = t; i < 16 * H_; i += 128) s_hb[i] = (bf16)h[base + i];
  for (int i = t; i < HEADS * KPH * HPH; i += 128) {
    int hd = i >> 10, rem = i & 1023, in = rem >> 5, o = rem & 31;
    int src = hd * 1024 + o * 32 + in;
    s_wqt[i] = (bf16)wq_l[src];
    s_wkt[i] = (bf16)wk_l[src];
    s_wvt[i] = (bf16)wv_l[src];
  }
  __syncthreads();
  // 8 tiles per projection: head(4) x ntile(2); per-head A slice is 16x32 at col hd*32
  for (int tt = wave; tt < 8; tt += 4) {
    int hd = tt >> 1, nt = tt & 1;
    v16bf a = load_A_bf(s_hb + hd * 32, H_, lane);
    v16bf bm = load_B_bf(s_wqt + hd * KPH * HPH, HPH, 0, nt * 16, lane);
    v8f acc = {};
    acc = WMMA_BF16(a, bm, acc);
    int col = hd * 32 + nt * 16 + (lane & 15);
    int mb = (lane & 16) ? 8 : 0;
#pragma unroll
    for (int r = 0; r < 8; ++r) qg[base + (size_t)(mb + r) * H_ + col] = acc[r];
  }
  for (int tt = wave; tt < 8; tt += 4) {
    int hd = tt >> 1, nt = tt & 1;
    v16bf a = load_A_bf(s_hb + hd * 32, H_, lane);
    v16bf bm = load_B_bf(s_wkt + hd * KPH * HPH, HPH, 0, nt * 16, lane);
    v8f acc = {};
    acc = WMMA_BF16(a, bm, acc);
    int col = hd * 32 + nt * 16 + (lane & 15);
    int mb = (lane & 16) ? 8 : 0;
#pragma unroll
    for (int r = 0; r < 8; ++r) kx[base + (size_t)(mb + r) * H_ + col] = (bf16)acc[r];
  }
  for (int tt = wave; tt < 8; tt += 4) {
    int hd = tt >> 1, nt = tt & 1;
    v16bf a = load_A_bf(s_hb + hd * 32, H_, lane);
    v16bf bm = load_B_bf(s_wvt + hd * HPH * HPH, HPH, 0, nt * 16, lane);
    v8f acc = {};
    acc = WMMA_BF16(a, bm, acc);
    int col = hd * 32 + nt * 16 + (lane & 15);
    int mb = (lane & 16) ? 8 : 0;
#pragma unroll
    for (int r = 0; r < 8; ++r) vx[base + (size_t)(mb + r) * H_ + col] = (bf16)acc[r];
  }
}

// GEMM1 (edge[48x128] @ w1^T -> ssp, stored bf16) then GEMM2 ( @ w2^T ) -> s_wk[48x32] f32
__device__ __forceinline__ void edge_mlp_path(const bf16* s_edge, const bf16* s_w1t,
                                              const float* s_b1, const bf16* s_w2t,
                                              const float* s_b2, bf16* s_t1, float* s_wk,
                                              int lane, int wave) {
  for (int tt = wave; tt < 6; tt += 4) {
    int mt = tt >> 1, nt = tt & 1;
    v8f acc = {};
#pragma unroll
    for (int ks = 0; ks < 4; ++ks) {
      v16bf a = load_A_bf(s_edge + mt * 16 * EDGE_CH + ks * 32, EDGE_CH, lane);
      v16bf bm = load_B_bf(s_w1t, KPH, ks * 32, nt * 16, lane);
      acc = WMMA_BF16(a, bm, acc);
    }
    int col = nt * 16 + (lane & 15);
    int mb = mt * 16 + ((lane & 16) ? 8 : 0);
#pragma unroll
    for (int r = 0; r < 8; ++r) s_t1[(mb + r) * KPH + col] = (bf16)sspf(acc[r] + s_b1[col]);
  }
  __syncthreads();
  for (int tt = wave; tt < 6; tt += 4) {
    int mt = tt >> 1, nt = tt & 1;
    v16bf a = load_A_bf(s_t1 + mt * 16 * KPH, KPH, lane);
    v16bf bm = load_B_bf(s_w2t, KPH, 0, nt * 16, lane);
    v8f acc = {};
    acc = WMMA_BF16(a, bm, acc);
    int col = nt * 16 + (lane & 15);
    int mb = mt * 16 + ((lane & 16) ? 8 : 0);
#pragma unroll
    for (int r = 0; r < 8; ++r) s_wk[(mb + r) * KPH + col] = acc[r] + s_b2[col];
  }
  __syncthreads();
}

__global__ __launch_bounds__(128) void attn_kernel(
    const unsigned short* __restrict__ edge_u16, const int* __restrict__ nbr,
    const unsigned short* __restrict__ kx_u16, const unsigned short* __restrict__ vx_u16,
    const float* __restrict__ qg, const float* __restrict__ wkn1,
    const float* __restrict__ bkn1, const float* __restrict__ wkn2,
    const float* __restrict__ bkn2, const float* __restrict__ wkl,
    const float* __restrict__ wvn1, const float* __restrict__ bvn1,
    const float* __restrict__ wvn2, const float* __restrict__ bvn2,
    const float* __restrict__ wvl, const float* __restrict__ bvl,
    float* __restrict__ aggr_out) {
  __shared__ bf16 s_w1t[EDGE_CH * KPH];  // transposed [in][out]
  __shared__ bf16 s_w2t[KPH * KPH];
  __shared__ bf16 s_wlt[KPH * KPH];
  __shared__ float s_b1[KPH], s_b2[KPH], s_bl[KPH];
  __shared__ __align__(16) bf16 s_edge[K_ * EDGE_CH];
  __shared__ float s_q[HEADS * KPH];
  __shared__ bf16 s_t1[K_ * KPH];
  __shared__ float s_wk[K_ * KPH];
  __shared__ bf16 s_ga[K_ * HEADS * KPH];
  __shared__ float s_logit[K_ * HEADS];
  __shared__ float s_alpha[K_ * HEADS];
  __shared__ float s_aggr[H_];
  __shared__ int s_nbr[K_];

  const bf16* edge = reinterpret_cast<const bf16*>(edge_u16);
  const bf16* kx = reinterpret_cast<const bf16*>(kx_u16);
  const bf16* vx = reinterpret_cast<const bf16*>(vx_u16);

  int bn = blockIdx.x, t = threadIdx.x;
  int lane = t & 31, wave = t >> 5;
  int bbase = (bn / N_) * N_;

  // ---- async direct-to-LDS staging of this node's 48x128 bf16 edge tile (12KB).
  // 6 x B128 per lane covers it; tracked by ASYNCcnt, drained before the barrier.
  {
    unsigned ldsa = (unsigned)(size_t)(void*)s_edge;
    unsigned long long ga =
        (unsigned long long)(const void*)(edge + (size_t)bn * K_ * EDGE_CH);
#pragma unroll
    for (int i = 0; i < 6; ++i) {
      unsigned off = (unsigned)(t * 16 + i * 2048);
      unsigned dl = ldsa + off;
      unsigned long long g = ga + off;
      asm volatile("global_load_async_to_lds_b128 %0, %1, off" : : "v"(dl), "v"(g) : "memory");
    }
    asm volatile("s_wait_asynccnt 0x0" : : : "memory");
  }

  // ---- stage q, neighbor list, k-path weights (transposed) ----
  s_q[t] = qg[(size_t)bn * H_ + t];
  if (t < K_) s_nbr[t] = nbr[bn * K_ + t];
  for (int i = t; i < EDGE_CH * KPH; i += 128) {
    int in = i >> 5, o = i & 31;
    s_w1t[i] = (bf16)wkn1[o * EDGE_CH + in];
  }
  for (int i = t; i < KPH * KPH; i += 128) {
    int in = i >> 5, o = i & 31;
    s_w2t[i] = (bf16)wkn2[o * KPH + in];
    s_wlt[i] = (bf16)wkl[o * KPH + in];
  }
  if (t < KPH) { s_b1[t] = bkn1[t]; s_b2[t] = bkn2[t]; }
  for (int i = t; i < K_ * HEADS; i += 128) s_logit[i] = 0.f;
  __syncthreads();

  // ---- k-path edge MLP: s_wk = w_k[48][32] ----
  edge_mlp_path(s_edge, s_w1t, s_b1, s_w2t, s_b2, s_t1, s_wk, lane, wave);

  // ---- gather neighbor k-features, scale by w_k ----
  for (int i = t; i < K_ * HEADS * KPH; i += 128) {
    int d = i & 31, r = i >> 5;
    int k = r >> 2, hd = r & 3;
    int j = s_nbr[k];
    float kv = (float)kx[(size_t)(bbase + j) * H_ + hd * KPH + d];
    s_ga[i] = (bf16)(s_wk[k * KPH + d] * kv);
  }
  __syncthreads();

  // ---- GEMM3 (wkl) fused with q.keys reduction into s_logit via LDS atomics.
  // The wkl bias contributes a per-head constant to every logit -> cancels in softmax.
  for (int tt = wave; tt < 24; tt += 4) {
    int mt = tt >> 1, nt = tt & 1;
    v16bf a = load_A_bf(s_ga + mt * 16 * KPH, KPH, lane);
    v16bf bm = load_B_bf(s_wlt, KPH, 0, nt * 16, lane);
    v8f acc = {};
    acc = WMMA_BF16(a, bm, acc);
    int col = nt * 16 + (lane & 15);
    int mb = mt * 16 + ((lane & 16) ? 8 : 0);
#pragma unroll
    for (int r = 0; r < 8; ++r) {
      int m = mb + r;  // row = k*HEADS + head
      atomicAdd(&s_logit[m], s_q[(m & 3) * KPH + col] * acc[r]);
    }
  }
  __syncthreads();

  // ---- softmax over the 48 neighbors, per head ----
  if (t < HEADS) {
    float mx = -3.0e38f;
    for (int k = 0; k < K_; ++k) mx = fmaxf(mx, s_logit[k * HEADS + t]);
    float sum = 0.f;
    for (int k = 0; k < K_; ++k) {
      float e = __expf(s_logit[k * HEADS + t] - mx);
      s_alpha[k * HEADS + t] = e;
      sum += e;
    }
    float inv = 1.f / sum;
    for (int k = 0; k < K_; ++k) s_alpha[k * HEADS + t] *= inv;
  }
  __syncthreads();

  // ---- re-stage v-path weights (reuse buffers) ----
  for (int i = t; i < EDGE_CH * KPH; i += 128) {
    int in = i >> 5, o = i & 31;
    s_w1t[i] = (bf16)wvn1[o * EDGE_CH + in];
  }
  for (int i = t; i < KPH * KPH; i += 128) {
    int in = i >> 5, o = i & 31;
    s_w2t[i] = (bf16)wvn2[o * KPH + in];
    s_wlt[i] = (bf16)wvl[o * KPH + in];
  }
  if (t < KPH) { s_b1[t] = bvn1[t]; s_b2[t] = bvn2[t]; s_bl[t] = bvl[t]; }
  s_aggr[t] = 0.f;
  __syncthreads();

  // ---- v-path edge MLP: s_wk = w_v[48][32] ----
  edge_mlp_path(s_edge, s_w1t, s_b1, s_w2t, s_b2, s_t1, s_wk, lane, wave);

  for (int i = t; i < K_ * HEADS * KPH; i += 128) {
    int d = i & 31, r = i >> 5;
    int k = r >> 2, hd = r & 3;
    int j = s_nbr[k];
    float vv = (float)vx[(size_t)(bbase + j) * H_ + hd * HPH + d];
    s_ga[i] = (bf16)(s_wk[k * KPH + d] * vv);
  }
  __syncthreads();

  // ---- GEMM3 (wvl) fused with alpha-weighted aggregation.
  // Sum_k alpha = 1, so the wvl bias folds into a single +bvl[d] at the end.
  for (int tt = wave; tt < 24; tt += 4) {
    int mt = tt >> 1, nt = tt & 1;
    v16bf a = load_A_bf(s_ga + mt * 16 * KPH, KPH, lane);
    v16bf bm = load_B_bf(s_wlt, KPH, 0, nt * 16, lane);
    v8f acc = {};
    acc = WMMA_BF16(a, bm, acc);
    int col = nt * 16 + (lane & 15);
    int mb = mt * 16 + ((lane & 16) ? 8 : 0);
#pragma unroll
    for (int r = 0; r < 8; ++r) {
      int m = mb + r;
      atomicAdd(&s_aggr[(m & 3) * HPH + col], s_alpha[m] * acc[r]);
    }
  }
  __syncthreads();

  aggr_out[(size_t)bn * H_ + t] = s_aggr[t] + s_bl[t & 31];
}

// Node update for 16 nodes per block: two 16x128x128 WMMA GEMMs + fused layernorm.
__global__ __launch_bounds__(128) void node_update16_kernel(
    float* __restrict__ h, const float* __restrict__ aggr, const float* __restrict__ cen_w,
    const float* __restrict__ cen_b, const float* __restrict__ out_w,
    const float* __restrict__ out_b, const float* __restrict__ ln_g,
    const float* __restrict__ ln_b) {
  __shared__ float s_hf[16 * H_];
  __shared__ bf16 s_hb[16 * H_];
  __shared__ bf16 s_wt[H_ * H_];  // transposed [in][out]
  __shared__ float s_x[16 * H_];
  __shared__ bf16 s_sx[16 * H_];
  __shared__ float s_red[128];
  int t = threadIdx.x, lane = t & 31, wave = t >> 5;
  size_t base = (size_t)blockIdx.x * 16 * H_;

  for (int i = t; i < 16 * H_; i += 128) {
    float v = h[base + i];
    s_hf[i] = v;
    s_hb[i] = (bf16)v;
  }
  for (int i = t; i < H_ * H_; i += 128) {
    int in = i >> 7, o = i & 127;
    s_wt[i] = (bf16)cen_w[o * H_ + in];
  }
  __syncthreads();

  // x = h @ cen_w^T + cen_b + aggr   (M=16, N=128, K=128)
  for (int nt = wave; nt < 8; nt += 4) {
    v8f acc = {};
#pragma unroll
    for (int ks = 0; ks < 4; ++ks) {
      v16bf a = load_A_bf(s_hb + ks * 32, H_, lane);
      v16bf bm = load_B_bf(s_wt, H_, ks * 32, nt * 16, lane);
      acc = WMMA_BF16(a, bm, acc);
    }
    int col = nt * 16 + (lane & 15);
    int mb = (lane & 16) ? 8 : 0;
#pragma unroll
    for (int r = 0; r < 8; ++r) {
      int m = mb + r;
      s_x[m * H_ + col] = acc[r] + cen_b[col] + aggr[base + (size_t)m * H_ + col];
    }
  }
  __syncthreads();

  // stage out_w (transposed) while doing layernorm; rows = nodes, 8 threads per row
  for (int i = t; i < H_ * H_; i += 128) {
    int in = i >> 7, o = i & 127;
    s_wt[i] = (bf16)out_w[o * H_ + in];
  }
  int row = t >> 3, l8 = t & 7;
  float p = 0.f;
  for (int c = l8; c < H_; c += 8) p += s_x[row * H_ + c];
  s_red[t] = p;
  __syncthreads();
  float mu = 0.f;
#pragma unroll
  for (int i = 0; i < 8; ++i) mu += s_red[row * 8 + i];
  mu *= (1.0f / H_);
  __syncthreads();
  float pv = 0.f;
  for (int c = l8; c < H_; c += 8) {
    float d = s_x[row * H_ + c] - mu;
    pv += d * d;
  }
  s_red[t] = pv;
  __syncthreads();
  float var = 0.f;
#pragma unroll
  for (int i = 0; i < 8; ++i) var += s_red[row * 8 + i];
  var *= (1.0f / H_);
  float rs = rsqrtf(var + 1e-5f);
  for (int c = l8; c < H_; c += 8) {
    float xn = (s_x[row * H_ + c] - mu) * rs * ln_g[c] + ln_b[c];
    s_sx[row * H_ + c] = (bf16)sspf(xn);
  }
  __syncthreads();

  // h += ssp(x) @ out_w^T + out_b
  for (int nt = wave; nt < 8; nt += 4) {
    v8f acc = {};
#pragma unroll
    for (int ks = 0; ks < 4; ++ks) {
      v16bf a = load_A_bf(s_sx + ks * 32, H_, lane);
      v16bf bm = load_B_bf(s_wt, H_, ks * 32, nt * 16, lane);
      acc = WMMA_BF16(a, bm, acc);
    }
    int col = nt * 16 + (lane & 15);
    int mb = (lane & 16) ? 8 : 0;
#pragma unroll
    for (int r = 0; r < 8; ++r) {
      int m = mb + r;
      h[base + (size_t)m * H_ + col] = s_hf[m * H_ + col] + acc[r] + out_b[col];
    }
  }
}

__global__ void fc1_kernel(const float* __restrict__ h, const float* __restrict__ w,
                           const float* __restrict__ bias, float* __restrict__ out) {
  __shared__ float sred[256];
  int b = blockIdx.x >> 7, o = blockIdx.x & 127, t = threadIdx.x;
  const float* x = h + (size_t)b * FLAT;
  const float* wr = w + (size_t)o * FLAT;
  float acc = 0.f;
  for (int i = t; i < FLAT; i += 256) {
    __builtin_prefetch(wr + i + 4096, 0, 0);  // global_prefetch_b8: stream 46MB of weights
    acc += x[i] * wr[i];
  }
  sred[t] = acc;
  __syncthreads();
  for (int s = 128; s > 0; s >>= 1) {
    if (t < s) sred[t] += sred[t + s];
    __syncthreads();
  }
  if (t == 0) out[b * 128 + o] = fmaxf(sred[0] + bias[o], 0.f);
}

__global__ void head_kernel(const float* __restrict__ x1, const float* __restrict__ w2,
                            const float* __restrict__ b2, const float* __restrict__ w3,
                            const float* __restrict__ b3, float* __restrict__ out) {
  __shared__ float sred[64];
  int b = blockIdx.x, t = threadIdx.x;
  const float* x = x1 + b * 128;
  float acc = b2[t];
#pragma unroll 8
  for (int i = 0; i < 128; ++i) acc += x[i] * w2[t * 128 + i];
  sred[t] = fmaxf(acc, 0.f) * w3[t];
  __syncthreads();
  for (int s = 32; s > 0; s >>= 1) {
    if (t < s) sred[t] += sred[t + s];
    __syncthreads();
  }
  if (t == 0) out[b] = sred[0] + b3[0];
}

// ---------------- launch ----------------

extern "C" void kernel_launch(void* const* d_in, const int* in_sizes, int n_in, void* d_out,
                              int out_size, void* d_ws, size_t ws_size, hipStream_t stream) {
  (void)in_sizes; (void)n_in; (void)out_size; (void)ws_size;
  const float* state   = (const float*)d_in[0];
  const float* ppv     = (const float*)d_in[1];
  const float* w_prot  = (const float*)d_in[2];
  const float* b_prot  = (const float*)d_in[3];
  const float* w_lig   = (const float*)d_in[4];
  const float* b_lig   = (const float*)d_in[5];
  const float* wq      = (const float*)d_in[6];
  const float* wk      = (const float*)d_in[7];
  const float* wv      = (const float*)d_in[8];
  const float* wkn1_w  = (const float*)d_in[9];
  const float* wkn1_b  = (const float*)d_in[10];
  const float* wkn2_w  = (const float*)d_in[11];
  const float* wkn2_b  = (const float*)d_in[12];
  const float* wkl_w   = (const float*)d_in[13];
  const float* wkl_b   = (const float*)d_in[14];
  const float* wvn1_w  = (const float*)d_in[15];
  const float* wvn1_b  = (const float*)d_in[16];
  const float* wvn2_w  = (const float*)d_in[17];
  const float* wvn2_b  = (const float*)d_in[18];
  const float* wvl_w   = (const float*)d_in[19];
  const float* wvl_b   = (const float*)d_in[20];
  const float* cen_w   = (const float*)d_in[21];
  const float* cen_b   = (const float*)d_in[22];
  const float* out_w   = (const float*)d_in[23];
  const float* out_b   = (const float*)d_in[24];
  const float* ln_g    = (const float*)d_in[25];
  const float* ln_b    = (const float*)d_in[26];
  const float* fc1_w   = (const float*)d_in[27];
  const float* fc1_b   = (const float*)d_in[28];
  const float* fc2_w   = (const float*)d_in[29];
  const float* fc2_b   = (const float*)d_in[30];
  const float* fc3_w   = (const float*)d_in[31];
  const float* fc3_b   = (const float*)d_in[32];
  (void)wkl_b;  // per-head constant in logits; cancels in softmax

  char* ws = (char*)d_ws;
  size_t off = 0;
  auto walloc = [&](size_t bytes) -> void* {
    void* p = ws + off;
    off = (off + bytes + 255) & ~(size_t)255;
    return p;
  };
  float*          h_buf = (float*)walloc((size_t)B_ * N_ * H_ * 4);
  float*          pos   = (float*)walloc((size_t)B_ * N_ * 3 * 4);
  int*            nbr   = (int*)walloc((size_t)B_ * N_ * K_ * 4);
  float*          dist  = (float*)walloc((size_t)B_ * N_ * K_ * 4);
  unsigned short* edge  = (unsigned short*)walloc((size_t)B_ * N_ * K_ * EDGE_CH * 2);
  unsigned short* kx    = (unsigned short*)walloc((size_t)B_ * N_ * H_ * 2);
  unsigned short* vx    = (unsigned short*)walloc((size_t)B_ * N_ * H_ * 2);
  float*          qg    = (float*)walloc((size_t)B_ * N_ * H_ * 4);
  float*          aggr  = (float*)walloc((size_t)B_ * N_ * H_ * 4);
  float*          x1    = (float*)walloc((size_t)B_ * 128 * 4);

  dim3 bnodes(B_ * N_);
  dim3 b16(B_ * N_ / 16);
  embed_kernel<<<bnodes, 128, 0, stream>>>(state, ppv, w_prot, b_prot, w_lig, b_lig, h_buf, pos);
  knn_kernel<<<bnodes, 128, 0, stream>>>(pos, nbr, dist);
  edge_kernel<<<bnodes, 128, 0, stream>>>(dist, edge);

  for (int l = 0; l < L_; ++l) {
    qkv16_kernel<<<b16, 128, 0, stream>>>(h_buf, wq + (size_t)l * HEADS * KPH * HPH,
                                          wk + (size_t)l * HEADS * KPH * HPH,
                                          wv + (size_t)l * HEADS * HPH * HPH, qg, kx, vx);
    attn_kernel<<<bnodes, 128, 0, stream>>>(
        edge, nbr, kx, vx, qg, wkn1_w + (size_t)l * KPH * EDGE_CH, wkn1_b + l * KPH,
        wkn2_w + (size_t)l * KPH * KPH, wkn2_b + l * KPH, wkl_w + (size_t)l * KPH * KPH,
        wvn1_w + (size_t)l * HPH * EDGE_CH, wvn1_b + l * HPH, wvn2_w + (size_t)l * HPH * HPH,
        wvn2_b + l * HPH, wvl_w + (size_t)l * HPH * HPH, wvl_b + l * HPH, aggr);
    node_update16_kernel<<<b16, 128, 0, stream>>>(h_buf, aggr, cen_w + (size_t)l * H_ * H_,
                                                  cen_b + l * H_, out_w + (size_t)l * H_ * H_,
                                                  out_b + l * H_, ln_g + l * H_, ln_b + l * H_);
  }

  fc1_kernel<<<dim3(B_ * 128), 256, 0, stream>>>(h_buf, fc1_w, fc1_b, x1);
  head_kernel<<<dim3(B_), 64, 0, stream>>>(x1, fc2_w, fc2_b, fc3_w, fc3_b, (float*)d_out);
}
